// NeuralCA_12652973654283
// MI455X (gfx1250) — matrix operand order repaired and checked by reading, not compiled, not run on hardware
//
#include <hip/hip_runtime.h>

typedef __attribute__((ext_vector_type(16))) __bf16 v16bf;
typedef __attribute__((ext_vector_type(8)))  __bf16 v8bf;
typedef __attribute__((ext_vector_type(8)))  float  v8f;

#define CCH   16
#define HID   128
#define HW    128
#define PLANE (HW*HW)          // 16384
#define CHW   (CCH*PLANE)      // 262144
#define THR   0.1f

// ---- dynamic LDS layout (bytes) ----
// xs   : float  [16][18][18]            20736
// hs   : __bf16 [4][16][136]            17408  (pitch 136 -> 16B-aligned rows)
// xnew : float  [16][16][16]            16384
// w1f  : __bf16 [8][2][32][16]          16384  (B-fragment layout, 32B/lane)
// w2f  : __bf16 [4][32][16]              4096  (K-permuted B-fragment layout)
#define XS_OFF   0
#define HS_OFF   20736
#define XN_OFF   (20736 + 17408)                    // 38144
#define W1F_OFF  (20736 + 17408 + 16384)            // 54528 (32B aligned)
#define W2F_OFF  (20736 + 17408 + 16384 + 16384)    // 70912 (32B aligned)
#define SMEM_BYTES (20736 + 17408 + 16384 + 16384 + 4096)   // 75008

#define HS_PITCH 136

// Hidden-unit permutation: h unit o = 16n + m (n = N-tile, m = column lane)
// is stored at hs slot  s = m*8 + n  -> one lane's 8 N-tile outputs are
// contiguous (enables packed b128 stores). GEMM2 uses the same permutation
// on w2's K index:  unit(s) = 16*(s&7) + (s>>3).

// ---------------------------------------------------------------------------
// Step kernel: x_mid = x + MLP(sobel(x)) for one 16x16 tile of one image.
// Writes unmasked x_mid to frame t+1, pre_life mask + unmasked ch0 to ws.
// ---------------------------------------------------------------------------
__global__ __launch_bounds__(128) void ca_step(
    float* __restrict__ out, int t, int S,
    const float* __restrict__ w1, const float* __restrict__ b1,
    const float* __restrict__ w2,
    float* __restrict__ ws_ch0, float* __restrict__ ws_pre)
{
  extern __shared__ unsigned char smem[];
  float*  xs   = (float*)(smem + XS_OFF);     // [c][18][18] linear
  __bf16* hs   = (__bf16*)(smem + HS_OFF);    // [wave][16][HS_PITCH]
  float*  xnew = (float*)(smem + XN_OFF);     // [c][16][16]
  __bf16* w1f  = (__bf16*)(smem + W1F_OFF);   // [(n*2+q)*32+lane][16]
  __bf16* w2f  = (__bf16*)(smem + W2F_OFF);   // [q*32+lane][16]

  const int tid  = threadIdx.x;
  const int lane = tid & 31;
  const int wave = tid >> 5;
  const int m16  = lane & 15;
  const int hi   = lane >> 4;            // 0: lanes 0-15, 1: lanes 16-31

  const int b   = blockIdx.z;
  const int gx0 = blockIdx.x * 16;
  const int gy0 = blockIdx.y * 16;

  const float* fin  = out + (size_t)(b*(S+1) + t    ) * CHW;
  float*       fout = out + (size_t)(b*(S+1) + t + 1) * CHW;

  // ---- stage x tile (wrap halo) into LDS via async-to-LDS path ----
  {
    const unsigned lds_base = (unsigned)(size_t)xs;   // low 32 bits = LDS addr
    for (int idx = tid; idx < CCH*18*18; idx += 128) {
      int c  = idx / 324;
      int r  = idx - c*324;
      int ry = r / 18;
      int rx = r - ry*18;
      int gy = (gy0 + ry - 1) & (HW-1);
      int gx = (gx0 + rx - 1) & (HW-1);
      unsigned lds_addr = lds_base + (unsigned)idx * 4u;
      int      goff     = (c*PLANE + gy*HW + gx) * 4;
      asm volatile("global_load_async_to_lds_b32 %0, %1, %2"
                   :: "v"(lds_addr), "v"(goff), "s"(fin) : "memory");
    }
  }

  // ---- build bf16 weight fragments in LDS (wave 0; per-lane layout) ----
  // B layout (32x16 K x N, bf16): lane l holds column n = l%16;
// element j = B[K = (j<8 ? j : j+8) + (l<16 ? 0 : 8)][n].
  if (wave == 0) {
    const int kb = hi * 8;
#pragma unroll
    for (int n = 0; n < 8; n++) {
#pragma unroll
      for (int q = 0; q < 2; q++) {
        __bf16* dst = w1f + (size_t)((n*2 + q)*32 + lane)*16;
#pragma unroll
        for (int j = 0; j < 16; j++) {
          int k   = q*32 + kb + ((j < 8) ? j : j + 8);
          int col = n*16 + m16;
          dst[j] = (__bf16)((k < 48) ? w1[col*48 + k] : 0.0f);
        }
      }
    }
    // w2 fragments with the K permutation applied
#pragma unroll
    for (int q = 0; q < 4; q++) {
      __bf16* dst = w2f + (size_t)(q*32 + lane)*16;
#pragma unroll
      for (int j = 0; j < 16; j++) {
        int kslot = q*32 + kb + ((j < 8) ? j : j + 8);                   // 0..127 (permuted K)
        int unit  = 16*(kslot & 7) + (kslot >> 3);   // original hidden unit
        dst[j] = (__bf16)w2[m16*HID + unit];
      }
    }
  }

  // bias values for the C operand of GEMM1 (per-lane column)
  float b1v[8];
#pragma unroll
  for (int n = 0; n < 8; n++) b1v[n] = b1[n*16 + m16];

  asm volatile("s_wait_asynccnt 0" ::: "memory");
  __syncthreads();

  // GEMM2 B-fragments are loop-invariant: hold in registers (4 x 8 VGPRs)
  v16bf b2f[4];
#pragma unroll
  for (int q = 0; q < 4; q++)
    b2f[q] = *(const v16bf*)(const void*)(w2f + (size_t)(q*32 + lane)*16);

  // ---- 4 M-tiles (tile rows) per wave; waves run decoupled (hs is
  //      per-wave private, only intra-wave LDS ordering is required) ----
  for (int j4 = 0; j4 < 4; j4++) {
    const int tt = wave*4 + j4;          // tile row 0..15 == M-tile index
    const int ly = tt + 1;
    const int lx = m16 + 1;

    // pre_life for this row's 16 pixels (lanes 0-15) -> straight to ws
    if (lane < 16) {
      float mx = -1e30f;
#pragma unroll
      for (int dy = -1; dy <= 1; dy++)
#pragma unroll
        for (int dx = -1; dx <= 1; dx++)
          mx = fmaxf(mx, xs[(ly+dy)*18 + (lane+1+dx)]);
      ws_pre[b*PLANE + (gy0+tt)*HW + (gx0+lane)] = (mx > THR) ? 1.0f : 0.0f;
    }

    // A fragments (16-bit A layout): lane l row m = l%16;
    // element i -> K = (i<8 ? i : i+8) + (l<16 ? 0 : 8) within 32-wide chunk.
    // Feature k = 3*c + f, f in {identity, sobel_x, sobel_y}.
    v16bf a0, a1;
#pragma unroll
    for (int i = 0; i < 16; i++) {
      int k = ((i < 8) ? i : i + 8) + hi*8;   // 0..31
      int c = k / 3, f = k - 3*c;
      const float* xc = xs + c*324;
      float d;
      if (f == 0)
        d = xc[ly*18 + lx];
      else if (f == 1)
        d = ( -     xc[(ly-1)*18 + lx-1] +      xc[(ly-1)*18 + lx+1]
              - 2.f*xc[(ly  )*18 + lx-1] + 2.f*xc[(ly  )*18 + lx+1]
              -     xc[(ly+1)*18 + lx-1] +      xc[(ly+1)*18 + lx+1]) * 0.125f;
      else
        d = (       xc[(ly-1)*18 + lx-1] + 2.f*xc[(ly-1)*18 + lx] + xc[(ly-1)*18 + lx+1]
              -     xc[(ly+1)*18 + lx-1] - 2.f*xc[(ly+1)*18 + lx] - xc[(ly+1)*18 + lx+1]) * 0.125f;
      a0[i] = (__bf16)d;
    }
#pragma unroll
    for (int i = 0; i < 16; i++) {
      float d = 0.0f;
      if (i < 8) {                        // K = 32..47 valid, rest zero pad
        int k = 32 + i + hi*8;
        int c = k / 3, f = k - 3*c;
        const float* xc = xs + c*324;
        if (f == 0)
          d = xc[ly*18 + lx];
        else if (f == 1)
          d = ( -     xc[(ly-1)*18 + lx-1] +      xc[(ly-1)*18 + lx+1]
                - 2.f*xc[(ly  )*18 + lx-1] + 2.f*xc[(ly  )*18 + lx+1]
                -     xc[(ly+1)*18 + lx-1] +      xc[(ly+1)*18 + lx+1]) * 0.125f;
        else
          d = (       xc[(ly-1)*18 + lx-1] + 2.f*xc[(ly-1)*18 + lx] + xc[(ly-1)*18 + lx+1]
                -     xc[(ly+1)*18 + lx-1] - 2.f*xc[(ly+1)*18 + lx] - xc[(ly+1)*18 + lx+1]) * 0.125f;
      }
      a1[i] = (__bf16)d;
    }

    __bf16* hw = hs + (size_t)wave*16*HS_PITCH;

    // ---- GEMM1: h = relu(delta @ w1^T + b1) ----
    // 8 independent accumulator chains: 16 WMMAs issue back-to-back with
    // reuse distance 8 (>= 5-slot bf16 hazard window) -> no NOP bubbles.
    v8f acc[8];
#pragma unroll
    for (int n = 0; n < 8; n++)
#pragma unroll
      for (int r = 0; r < 8; r++) acc[n][r] = b1v[n];
#pragma unroll
    for (int n = 0; n < 8; n++) {
      v16bf bq = *(const v16bf*)(const void*)(w1f + (size_t)((n*2 + 0)*32 + lane)*16);
      acc[n] = __builtin_amdgcn_wmma_f32_16x16x32_bf16(false, a0, false, bq,
                                                       (short)0, acc[n], false, false);
    }
#pragma unroll
    for (int n = 0; n < 8; n++) {
      v16bf bq = *(const v16bf*)(const void*)(w1f + (size_t)((n*2 + 1)*32 + lane)*16);
      acc[n] = __builtin_amdgcn_wmma_f32_16x16x32_bf16(false, a1, false, bq,
                                                       (short)0, acc[n], false, false);
    }
    // relu + pack 8 N-tile values/row into one 16B LDS store (permuted slots)
#pragma unroll
    for (int r = 0; r < 8; r++) {
      v8bf pk;
#pragma unroll
      for (int n = 0; n < 8; n++) {
        float v = acc[n][r];
        v = (v > 0.0f) ? v : 0.0f;
        pk[n] = (__bf16)v;
      }
      *(v8bf*)(void*)(hw + (size_t)(r + hi*8)*HS_PITCH + m16*8) = pk;
    }

    // intra-wave cross-lane LDS ordering only: compiler barrier, no s_barrier
    __builtin_amdgcn_fence(__ATOMIC_ACQ_REL, "wavefront");

    // ---- GEMM2: delta2 = h @ w2^T (K permuted), two independent accs ----
    const __bf16* hrow = hw + (size_t)m16*HS_PITCH;
    v16bf aH[4];
#pragma unroll
    for (int q = 0; q < 4; q++) {
      int k1 = q*32 + hi*8;              // multiple of 8 -> 16B-aligned chunks
      v8bf lo  = *(const v8bf*)(const void*)(hrow + k1);
      v8bf hi8 = *(const v8bf*)(const void*)(hrow + k1 + 16);
      aH[q] = __builtin_shufflevector(lo, hi8, 0,1,2,3,4,5,6,7,
                                               8,9,10,11,12,13,14,15);
    }
    v8f zacc;
#pragma unroll
    for (int r = 0; r < 8; r++) zacc[r] = 0.0f;
    v8f acc2a = __builtin_amdgcn_wmma_f32_16x16x32_bf16(false, aH[0], false, b2f[0],
                                                        (short)0, zacc, false, false);
    v8f acc2b = __builtin_amdgcn_wmma_f32_16x16x32_bf16(false, aH[1], false, b2f[1],
                                                        (short)0, zacc, false, false);
    acc2a = __builtin_amdgcn_wmma_f32_16x16x32_bf16(false, aH[2], false, b2f[2],
                                                    (short)0, acc2a, false, false);
    acc2b = __builtin_amdgcn_wmma_f32_16x16x32_bf16(false, aH[3], false, b2f[3],
                                                    (short)0, acc2b, false, false);

    // D layout: lane l, vgpr r -> pixel (tt, r + hi*8), channel m16
#pragma unroll
    for (int r = 0; r < 8; r++) {
      int mp = r + hi*8;
      xnew[m16*256 + tt*16 + mp] =
          xs[m16*324 + (tt+1)*18 + (mp+1)] + (acc2a[r] + acc2b[r]);
    }
  }

  __syncthreads();

  // ---- coalesced flush of x_mid; stash channel-0 plane for mask kernel ----
  for (int idx = tid; idx < CCH*16*16; idx += 128) {
    int c  = idx >> 8;
    int r  = idx & 255;
    int yy = r >> 4;
    int xx = r & 15;
    float v = xnew[c*256 + yy*16 + xx];
    int gy = gy0 + yy, gx = gx0 + xx;
    fout[c*PLANE + gy*HW + gx] = v;
    if (c == 0) ws_ch0[b*PLANE + gy*HW + gx] = v;
  }
}

// ---------------------------------------------------------------------------
// Mask kernel: post_life = maxpool3x3(wrap) on unmasked x_mid ch0 (from ws),
// factor = pre & post, applied in place to frame t+1.
// ---------------------------------------------------------------------------
__global__ __launch_bounds__(256) void ca_mask(
    float* __restrict__ out, int t, int S,
    const float* __restrict__ ws_ch0, const float* __restrict__ ws_pre)
{
  int p = blockIdx.x * 256 + threadIdx.x;   // 0 .. 16*PLANE-1
  int b = p >> 14;
  int r = p & (PLANE-1);
  int y = r >> 7;
  int x = r & (HW-1);
  const float* c0 = ws_ch0 + b*PLANE;
  float mx = -1e30f;
#pragma unroll
  for (int dy = -1; dy <= 1; dy++) {
    int yy = (y + dy) & (HW-1);
#pragma unroll
    for (int dx = -1; dx <= 1; dx++) {
      int xx = (x + dx) & (HW-1);
      mx = fmaxf(mx, c0[yy*HW + xx]);
    }
  }
  float factor = ((mx > THR) && (ws_pre[p] > 0.5f)) ? 1.0f : 0.0f;
  float* fr = out + (size_t)(b*(S+1) + t + 1) * CHW + y*HW + x;
#pragma unroll
  for (int c = 0; c < CCH; c++)
    fr[c*PLANE] *= factor;
}

// ---------------------------------------------------------------------------
// Copy kernel: frame 0 of every batch image = input x
// ---------------------------------------------------------------------------
__global__ __launch_bounds__(256) void ca_copy(
    float* __restrict__ out, const float* __restrict__ x, int S)
{
  long i = (long)blockIdx.x * 256 + threadIdx.x;
  if (i < 16L*CHW) {
    long b = i / CHW;
    long r = i - b*CHW;
    out[(size_t)(b*(S+1))*CHW + r] = x[i];
  }
}

extern "C" void kernel_launch(void* const* d_in, const int* in_sizes, int n_in,
                              void* d_out, int out_size, void* d_ws, size_t ws_size,
                              hipStream_t stream)
{
  const float* x  = (const float*)d_in[0];
  const float* w1 = (const float*)d_in[1];
  const float* b1 = (const float*)d_in[2];
  const float* w2 = (const float*)d_in[3];
  (void)in_sizes; (void)n_in; (void)ws_size;

  // out_size = B*(S+1)*C*H*W with B=16, C*H*W = 262144
  int S = (int)((long)out_size / (16L * CHW)) - 1;

  float* out    = (float*)d_out;
  float* ws_ch0 = (float*)d_ws;               // 16*PLANE floats
  float* ws_pre = ws_ch0 + 16*PLANE;          // 16*PLANE floats (~2.1 MB total)

  // allow > 64KB dynamic LDS (gfx1250: up to 320KB per workgroup)
  (void)hipFuncSetAttribute((const void*)ca_step,
                            hipFuncAttributeMaxDynamicSharedMemorySize,
                            SMEM_BYTES);

  ca_copy<<<dim3((16*CHW + 255)/256), 256, 0, stream>>>(out, x, S);
  for (int t = 0; t < S; t++) {
    ca_step<<<dim3(8, 8, 16), 128, SMEM_BYTES, stream>>>(out, t, S, w1, b1, w2,
                                                         ws_ch0, ws_pre);
    ca_mask<<<dim3(16*PLANE/256), 256, 0, stream>>>(out, t, S, ws_ch0, ws_pre);
  }
}
